// HGNN_45268955300429
// MI455X (gfx1250) — matrix operand order repaired and checked by reading, not compiled, not run on hardware
//
#include <hip/hip_runtime.h>
#include <hip/hip_bf16.h>

// ---------------------------------------------------------------------------
// HGNN for MI455X (gfx1250, wave32).
//  - Aggregations (memory-bound, ~10GB traffic): fused edge-MLP + gather +
//    global_atomic_add_f32 scatter, one wave per edge, float4 per lane.
//  - GEMMs: v_wmma_f32_16x16x32_bf16. Block = 128 rows x 128 cols, 8 waves.
//    Weights staged once per block into LDS as transposed bf16; each wave
//    preloads all K/32 B-fragments into registers, then unrolled row loop.
//  - BN: two-pass (atomic column sums, then normalize), 0.5 mean folded in.
// ---------------------------------------------------------------------------

typedef __bf16 bf16_t;
typedef __attribute__((ext_vector_type(16))) __bf16 v16bf;
typedef __attribute__((ext_vector_type(8)))  __bf16 v8bf;
typedef __attribute__((ext_vector_type(8)))  float  v8f;

#define NN   100000   // N0 == N1
#define EE   800000
#define SEQL 256
#define EMB  128

__device__ __forceinline__ void load8f(const float* __restrict__ p, float* o) {
    float4 a = *(const float4*)(p);
    float4 b = *(const float4*)(p + 4);
    o[0] = a.x; o[1] = a.y; o[2] = a.z; o[3] = a.w;
    o[4] = b.x; o[5] = b.y; o[6] = b.z; o[7] = b.w;
}

// C = relu?( (A_eff @ B + bias) * alpha + Cin ), A_eff = A + a2scale*A2.
// A: MxK row-major f32, B: KxN row-major f32, C: MxN row-major f32 (ldc=N).
// grid.x = ceil(M/128), grid.y = N/128, block = 256 (8 waves).
// Wave w owns column tile w of the block's 128-col strip; loops 8 row tiles.
template<int K, bool HASA2>
__global__ __launch_bounds__(256)
void gemm_wmma_kernel(const float* __restrict__ A, const float* __restrict__ A2,
                      float a2scale,
                      const float* __restrict__ B, const float* __restrict__ bias,
                      const float* __restrict__ Cin, float* __restrict__ Cout,
                      int M, int N, float alpha, int relu_flag)
{
    extern __shared__ __align__(16) char smem[];
    bf16_t* sBt = (bf16_t*)smem;                 // [128 cols][K] bf16, transposed
    const int colbase = blockIdx.y * 128;

    // Cooperative stage: B (f32, row-major) -> LDS bf16 transposed.
    for (int idx = threadIdx.x; idx < K * 128; idx += 256) {
        const int k = idx >> 7;
        const int c = idx & 127;
        sBt[c * K + k] = (bf16_t)B[(long)k * N + colbase + c];
    }
    __syncthreads();

    const int lane = threadIdx.x & 31;
    const int wave = threadIdx.x >> 5;
    const int half = lane >> 4;                  // lane half: K/row sub-group
    const int l15  = lane & 15;
    const int cl   = wave * 16 + l15;            // local column (0..127)
    const int bcol = colbase + cl;

    // Preload all B fragments from LDS into registers (2x ds_load_b128 each).
    // B frag layout: elems 0-7 -> K = k0 + 8*half + i, elems 8-15 -> K = k0 + 16 + 8*half + (i-8), N = bcol.
    constexpr int KT = K / 32;
    v16bf bfr[KT];
    #pragma unroll
    for (int kt = 0; kt < KT; ++kt) {
        const bf16_t* bp = sBt + cl * K + kt * 32 + 8 * half;
        v8bf lo = *(const v8bf*)bp;
        v8bf hi = *(const v8bf*)(bp + 16);
        bfr[kt] = __builtin_shufflevector(lo, hi,
                    0, 1, 2, 3, 4, 5, 6, 7, 8, 9, 10, 11, 12, 13, 14, 15);
    }

    const float bv = bias[bcol];
    #pragma unroll
    for (int mt = 0; mt < 8; ++mt) {
        const int row0 = (blockIdx.x * 8 + mt) * 16;
        if (row0 < M) {
            const long arow = row0 + l15;
            v8f acc = {};
            #pragma unroll
            for (int kt = 0; kt < KT; ++kt) {
                // A fragment: elems 0-7 -> K = 32*kt+8*half+[0..7],
                //             elems 8-15 -> K = 32*kt+16+8*half+[0..7]
                float fa[16];
                const float* ap = A + arow * (long)K + kt * 32 + 8 * half;
                load8f(ap, fa);
                load8f(ap + 16, fa + 8);
                if (HASA2) {
                    float fb[16];
                    const float* ap2 = A2 + arow * (long)K + kt * 32 + 8 * half;
                    load8f(ap2, fb);
                    load8f(ap2 + 16, fb + 8);
                    #pragma unroll
                    for (int i = 0; i < 16; ++i) fa[i] = fmaf(a2scale, fb[i], fa[i]);
                }
                v16bf a;
                #pragma unroll
                for (int i = 0; i < 16; ++i) a[i] = (bf16_t)fa[i];
                acc = __builtin_amdgcn_wmma_f32_16x16x32_bf16(
                          false, a, false, bfr[kt], (short)0, acc, false, false);
            }
            // C/D layout: col = lane&15, row = elem + 8*half
            #pragma unroll
            for (int v = 0; v < 8; ++v) {
                const long m = row0 + v + 8 * half;
                float val = (acc[v] + bv) * alpha;
                if (Cin) val += Cin[m * (long)N + bcol];
                if (relu_flag) val = fmaxf(val, 0.0f);
                Cout[m * (long)N + bcol] = val;
            }
        }
    }
}

// One wave per edge: m = h_src[src] + (ea[e] @ We + be); atomicAdd to agg[dst].
// Lane l covers features [4l, 4l+4). block = 256 -> 8 edges per block.
__global__ __launch_bounds__(256)
void agg_kernel(const float* __restrict__ h_src, const float* __restrict__ ea,
                const int* __restrict__ ei, const float* __restrict__ We,
                const float* __restrict__ be, float* __restrict__ agg, int E)
{
    const int e = (int)((blockIdx.x * (unsigned)blockDim.x + threadIdx.x) >> 5);
    if (e >= E) return;
    const int lane = threadIdx.x & 31;
    const int c0   = lane * 4;
    const int src  = ei[e];        // ei[0][e]
    const int dst  = ei[E + e];    // ei[1][e]

    float4 h  = *(const float4*)(h_src + (long)src * EMB + c0);
    float4 ew = *(const float4*)(be + c0);
    #pragma unroll
    for (int k = 0; k < 16; ++k) {
        const float  a = ea[(long)e * 16 + k];            // wave-uniform
        const float4 w = *(const float4*)(We + k * EMB + c0);
        ew.x = fmaf(a, w.x, ew.x);
        ew.y = fmaf(a, w.y, ew.y);
        ew.z = fmaf(a, w.z, ew.z);
        ew.w = fmaf(a, w.w, ew.w);
    }
    float* p = agg + (long)dst * EMB + c0;
    __hip_atomic_fetch_add(p + 0, h.x + ew.x, __ATOMIC_RELAXED, __HIP_MEMORY_SCOPE_AGENT);
    __hip_atomic_fetch_add(p + 1, h.y + ew.y, __ATOMIC_RELAXED, __HIP_MEMORY_SCOPE_AGENT);
    __hip_atomic_fetch_add(p + 2, h.z + ew.z, __ATOMIC_RELAXED, __HIP_MEMORY_SCOPE_AGENT);
    __hip_atomic_fetch_add(p + 3, h.w + ew.w, __ATOMIC_RELAXED, __HIP_MEMORY_SCOPE_AGENT);
}

#define BN_ROWS 256
// Column sums of (prescale * X): block of 128 threads (one per column),
// BN_ROWS rows per block, atomic partial-sum into sum[]/sq[].
__global__ __launch_bounds__(128)
void bn_stats_kernel(const float* __restrict__ X, float* __restrict__ sum,
                     float* __restrict__ sq, int M, float prescale)
{
    const int c  = threadIdx.x;
    const int r0 = blockIdx.x * BN_ROWS;
    const int r1 = (r0 + BN_ROWS < M) ? (r0 + BN_ROWS) : M;
    float s = 0.0f, q = 0.0f;
    for (int r = r0; r < r1; ++r) {
        const float x = X[(long)r * EMB + c] * prescale;
        s += x;
        q = fmaf(x, x, q);
    }
    __hip_atomic_fetch_add(&sum[c], s, __ATOMIC_RELAXED, __HIP_MEMORY_SCOPE_AGENT);
    __hip_atomic_fetch_add(&sq[c],  q, __ATOMIC_RELAXED, __HIP_MEMORY_SCOPE_AGENT);
}

__global__ __launch_bounds__(256)
void bn_norm_kernel(const float* __restrict__ X, const float* __restrict__ sum,
                    const float* __restrict__ sq, const float* __restrict__ gamma,
                    const float* __restrict__ beta, float* __restrict__ Y,
                    int M, float prescale, int relu_flag)
{
    const long i = (long)blockIdx.x * blockDim.x + threadIdx.x;
    if (i >= (long)M * EMB) return;
    const int   c    = (int)(i & (EMB - 1));
    const float invM = 1.0f / (float)M;
    const float mean = sum[c] * invM;
    const float var  = fmaxf(sq[c] * invM - mean * mean, 0.0f);
    const float x    = X[i] * prescale;
    float y = gamma[c] * (x - mean) * rsqrtf(var + 1e-5f) + beta[c];
    if (relu_flag) y = fmaxf(y, 0.0f);
    Y[i] = y;
}

extern "C" void kernel_launch(void* const* d_in, const int* in_sizes, int n_in,
                              void* d_out, int out_size, void* d_ws, size_t ws_size,
                              hipStream_t stream)
{
    (void)in_sizes; (void)n_in; (void)out_size; (void)ws_size;

    const float* x0   = (const float*)d_in[0];
    const float* x1   = (const float*)d_in[1];
    const float* ea11 = (const float*)d_in[2];
    const float* ea10 = (const float*)d_in[3];
    const float* ea01 = (const float*)d_in[4];
    const float* ea00 = (const float*)d_in[5];
    const int*   ei11 = (const int*)d_in[6];
    const int*   ei10 = (const int*)d_in[7];
    const int*   ei01 = (const int*)d_in[8];
    const int*   ei00 = (const int*)d_in[9];
    const float* Wx  = (const float*)d_in[10];
    const float* bx  = (const float*)d_in[11];
    const float* We  = (const float*)d_in[12];
    const float* be  = (const float*)d_in[13];
    const float* gW1 = (const float*)d_in[14];
    const float* gb1 = (const float*)d_in[15];
    const float* gW2 = (const float*)d_in[16];
    const float* gb2 = (const float*)d_in[17];
    const float* W10 = (const float*)d_in[18];
    const float* b10 = (const float*)d_in[19];
    const float* W01 = (const float*)d_in[20];
    const float* b01 = (const float*)d_in[21];
    const float* W00 = (const float*)d_in[22];
    const float* b00 = (const float*)d_in[23];
    const float* bn_g = (const float*)d_in[24];
    const float* bn_b = (const float*)d_in[25];

    const size_t NF = (size_t)NN * EMB;        // 12.8M floats per node buffer
    float* ws   = (float*)d_ws;
    float* h0   = ws;                          // node feats type '0'
    float* h1   = h0  + NF;                    // node feats type '1'
    float* s11  = h1  + NF;                    // GIN aggregation (1->1)
    float* a01  = s11 + NF;                    // agg (0->1)
    float* a10  = a01 + NF;                    // agg (1->0)
    float* a00  = a10 + NF;                    // agg (0->0)
    float* n0   = a00 + NF;                    // new h0 pre-BN (sum, not *0.5)
    float* n1   = n0  + NF;                    // new h1 pre-BN
    float* stats = n1 + NF;                    // 512 floats: sum0,sq0,sum1,sq1
    float* Tbuf = a01;                         // GIN hidden 100k x 256, overlays a01+a10

    const dim3 blk(256);
    const int  MB = (NN / 16 + 7) / 8;         // row-blocks of 128 rows (tail-safe)
    const dim3 g128(MB, 1);                    // N = 128
    const dim3 g256(MB, 2);                    // N = 256
    const size_t shm128 = 128 * 128 * sizeof(bf16_t);   // 32 KB
    const size_t shm256 = 128 * 256 * sizeof(bf16_t);   // 64 KB

    // Input embeddings: h = x @ Wx + bx  (K=256, N=128)
    gemm_wmma_kernel<256, false><<<g128, blk, shm256, stream>>>(
        x0, nullptr, 0.0f, Wx, bx, nullptr, h0, NN, EMB, 1.0f, 0);
    gemm_wmma_kernel<256, false><<<g128, blk, shm256, stream>>>(
        x1, nullptr, 0.0f, Wx, bx, nullptr, h1, NN, EMB, 1.0f, 0);

    for (int layer = 0; layer < 2; ++layer) {
        // zero the four contiguous aggregation buffers
        hipMemsetAsync(s11, 0, 4 * NF * sizeof(float), stream);

        const dim3 gagg(EE / 8);
        agg_kernel<<<gagg, blk, 0, stream>>>(h1, ea11, ei11, We, be, s11, EE);
        agg_kernel<<<gagg, blk, 0, stream>>>(h0, ea01, ei01, We, be, a01, EE);
        agg_kernel<<<gagg, blk, 0, stream>>>(h1, ea10, ei10, We, be, a10, EE);
        agg_kernel<<<gagg, blk, 0, stream>>>(h0, ea00, ei00, We, be, a00, EE);

        // n1 = (a01 @ W01 + b01) * 0.1
        gemm_wmma_kernel<128, false><<<g128, blk, shm128, stream>>>(
            a01, nullptr, 0.0f, W01, b01, nullptr, n1, NN, EMB, 0.1f, 0);
        // n0 = (a10 @ W10 + b10) * 0.1
        gemm_wmma_kernel<128, false><<<g128, blk, shm128, stream>>>(
            a10, nullptr, 0.0f, W10, b10, nullptr, n0, NN, EMB, 0.1f, 0);
        // n0 += (a00 @ W00 + b00) * 0.1
        gemm_wmma_kernel<128, false><<<g128, blk, shm128, stream>>>(
            a00, nullptr, 0.0f, W00, b00, n0, n0, NN, EMB, 0.1f, 0);
        // T = relu((s11 + 1.1*h1) @ gW1 + gb1)   (N=256; Tbuf overlays a01+a10)
        gemm_wmma_kernel<128, true><<<g256, blk, shm128, stream>>>(
            s11, h1, 1.1f, gW1, gb1, nullptr, Tbuf, NN, 2 * EMB, 1.0f, 1);
        // n1 += T @ gW2 + gb2
        gemm_wmma_kernel<256, false><<<g128, blk, shm256, stream>>>(
            Tbuf, nullptr, 0.0f, gW2, gb2, n1, n1, NN, EMB, 1.0f, 0);

        // BatchNorm (the 0.5 HeteroConv-mean is folded into prescale)
        hipMemsetAsync(stats, 0, 512 * sizeof(float), stream);
        const dim3 gstat((NN + BN_ROWS - 1) / BN_ROWS);
        bn_stats_kernel<<<gstat, dim3(128), 0, stream>>>(n0, stats + 0,   stats + 128, NN, 0.5f);
        bn_stats_kernel<<<gstat, dim3(128), 0, stream>>>(n1, stats + 256, stats + 384, NN, 0.5f);

        const int last = (layer == 1);
        float* d0 = last ? (float*)d_out        : h0;   // output = concat(h0, h1)
        float* d1 = last ? (float*)d_out + NF   : h1;
        const dim3 gnorm((unsigned)((NF + 255) / 256));
        bn_norm_kernel<<<gnorm, blk, 0, stream>>>(n0, stats + 0,   stats + 128,
                                                  bn_g + layer * EMB, bn_b + layer * EMB,
                                                  d0, NN, 0.5f, !last);
        bn_norm_kernel<<<gnorm, blk, 0, stream>>>(n1, stats + 256, stats + 384,
                                                  bn_g + layer * EMB, bn_b + layer * EMB,
                                                  d1, NN, 0.5f, !last);
    }
}
